// SummaryDecoder_20495583937101
// MI455X (gfx1250) — matrix operand order repaired and checked by reading, not compiled, not run on hardware
//
#include <hip/hip_runtime.h>
#include <hip/hip_bf16.h>
#include <cstdint>
#include <cstddef>

namespace {

constexpr int kB = 32;
constexpr int kL = 4096;
constexpr int kH = 512;
constexpr int kV = 32000;

typedef __bf16 v16bf __attribute__((ext_vector_type(16)));
typedef float  v8f   __attribute__((ext_vector_type(8)));
typedef int    v4i   __attribute__((vector_size(16)));   // matches builtin param type

#define AS1 __attribute__((address_space(1)))
#define AS3 __attribute__((address_space(3)))

__device__ __forceinline__ float sig(float x) { return 1.0f / (1.0f + __expf(-x)); }

__device__ __forceinline__ void wait_async0() {
#if __has_builtin(__builtin_amdgcn_s_wait_asynccnt)
  __builtin_amdgcn_s_wait_asynccnt(0);
#else
  asm volatile("s_wait_asynccnt 0x0" ::: "memory");
#endif
}

// ---------------- LSTM gates: gates = x@Wih^T + bih + h0@Whh^T + bhh ----------------
__global__ void gates_kernel(const float* __restrict__ x, const float* __restrict__ h0,
                             const float* __restrict__ Wih, const float* __restrict__ Whh,
                             const float* __restrict__ bih, const float* __restrict__ bhh,
                             float* __restrict__ gates) {
  int idx = blockIdx.x * blockDim.x + threadIdx.x;       // [0, 32*2048)
  int b = idx >> 11, j = idx & 2047;
  const float* xr = x + (size_t)b * kH;
  const float* hr = h0 + (size_t)b * kH;
  const float* wi = Wih + (size_t)j * kH;
  const float* wh = Whh + (size_t)j * kH;
  float acc = bih[j] + bhh[j];
  for (int k = 0; k < kH; ++k) acc = fmaf(xr[k], wi[k], fmaf(hr[k], wh[k], acc));
  gates[idx] = acc;
}

// ---------------- h, c from gates (torch order i,f,g,o) ----------------
__global__ void hc_kernel(const float* __restrict__ gates, const float* __restrict__ c0,
                          float* __restrict__ h_out, float* __restrict__ c_out) {
  int idx = blockIdx.x * blockDim.x + threadIdx.x;       // [0, 32*512)
  int b = idx >> 9, n = idx & 511;
  const float* g = gates + (size_t)b * 2048;
  float gi = g[n], gf = g[512 + n], gg = g[1024 + n], go = g[1536 + n];
  float c = sig(gf) * c0[idx] + sig(gi) * tanhf(gg);
  float h = sig(go) * tanhf(c);
  h_out[idx] = h;
  c_out[idx] = c;
}

// ---------------- qb2 = h@W1^T + W1_b + W2_b ----------------
__global__ void q_kernel(const float* __restrict__ h, const float* __restrict__ W1w,
                         const float* __restrict__ W1b, const float* __restrict__ W2b,
                         float* __restrict__ qb2) {
  int idx = blockIdx.x * blockDim.x + threadIdx.x;       // [0, 32*512)
  int b = idx >> 9, n = idx & 511;
  const float* hr = h + (size_t)b * kH;
  const float* wr = W1w + (size_t)n * kH;
  float acc = W1b[n];
  for (int k = 0; k < kH; ++k) acc = fmaf(hr[k], wr[k], acc);
  qb2[idx] = acc + W2b[n];
}

// ---------------- pack W2 into WMMA B-fragment layout (bf16) ----------------
// frag f = n_tile*16 + k_chunk ; element addr = f*512 + lane*16 + slot
// B lane mapping (16-bit 32x16): lane -> N = t*16 + (lane&15); slot -> K = kc*32 + (s>>3)*16 + (lane>>4)*8 + (s&7)
__global__ void pack_w2_kernel(const float* __restrict__ W2w, __bf16* __restrict__ frag) {
  int idx = blockIdx.x * blockDim.x + threadIdx.x;       // [0, 32*16*512)
  int s = idx & 15;
  int lane = (idx >> 4) & 31;
  int f = idx >> 9;
  int t = f >> 4, kc = f & 15;
  int N = t * 16 + (lane & 15);
  int K = kc * 32 + ((s >> 3) << 4) + ((lane >> 4) << 3) + (s & 7);
  frag[idx] = (__bf16)W2w[(size_t)N * kH + K];
}

// ---------------- fused scores kernel (WMMA bf16 + async-to-LDS B staging) ----------
// scores[b,l] = u1 . tanh(qb2[b] + doc[b,l] @ W2^T)
// one wave per 16-row M tile (4 tiles / block). doc tile staged f32->bf16 into LDS in
// A-fragment layout; W2 B-fragments staged per-block into LDS with async copies and
// shared by all 4 waves (4x less L2 traffic).
__global__ void __launch_bounds__(128)
scores_kernel(const float* __restrict__ doc, const __bf16* __restrict__ w2frag,
              const float* __restrict__ qb2, const float* __restrict__ u1,
              float* __restrict__ scores) {
  extern __shared__ char smem[];                  // 64KB A frags + 64KB B frags
  __bf16* Alds = (__bf16*)smem;                   // [wave][kc][lane*16+s] : 4*16*512
  __bf16* Blds = (__bf16*)(smem + 65536);         // [tt*16+kc][lane*16+s] : 64*512

  const int wave = threadIdx.x >> 5;
  const int lane = threadIdx.x & 31;
  const int nl = lane & 15, kh = lane >> 4;
  const int mtile = blockIdx.x * 4 + wave;        // 8192 tiles total
  const int row0 = mtile * 16;
  const int b = row0 >> 12;                       // row0 / 4096 (tiles never cross b)
  const int l0 = row0 & 4095;

  // ---- stage this wave's doc tile into LDS as 16 A-fragments (f32 -> bf16) ----
  // A lane mapping (16-bit 16x32): lane -> M = lane&15; slots 0..7 -> K = kc*32+kh*8+s,
  // slots 8..15 -> K = kc*32+16+kh*8+(s-8)
  {
    const float* rowp = doc + ((size_t)row0 + nl) * kH;
    for (int kc = 0; kc < 16; ++kc) {
      const float* rp = rowp + kc * 32 + kh * 8;
      __builtin_prefetch(rp + 32, 0, 1);          // global_prefetch_b8 on the doc stream
      float4 x0 = *(const float4*)rp;
      float4 x1 = *(const float4*)(rp + 4);
      float4 y0 = *(const float4*)(rp + 16);
      float4 y1 = *(const float4*)(rp + 20);
      __bf16* dst = Alds + ((size_t)wave * 16 + kc) * 512 + lane * 16;
      dst[0] = (__bf16)x0.x;  dst[1] = (__bf16)x0.y;  dst[2]  = (__bf16)x0.z;  dst[3]  = (__bf16)x0.w;
      dst[4] = (__bf16)x1.x;  dst[5] = (__bf16)x1.y;  dst[6]  = (__bf16)x1.z;  dst[7]  = (__bf16)x1.w;
      dst[8] = (__bf16)y0.x;  dst[9] = (__bf16)y0.y;  dst[10] = (__bf16)y0.z;  dst[11] = (__bf16)y0.w;
      dst[12] = (__bf16)y1.x; dst[13] = (__bf16)y1.y; dst[14] = (__bf16)y1.z;  dst[15] = (__bf16)y1.w;
    }
  }

  float part[8];
#pragma unroll
  for (int j = 0; j < 8; ++j) part[j] = 0.0f;

  // 8 N-super-tiles x 4 N-tiles = 32 N-tiles over H=512
  for (int tsup = 0; tsup < 8; ++tsup) {
    __syncthreads();                              // previous super-tile fully consumed
    // ---- stage 64KB of B fragments for this super-tile into LDS (block-shared) ----
    {
      const char* src = (const char*)(w2frag + (size_t)tsup * 64 * 512);
#if __has_builtin(__builtin_amdgcn_global_load_async_to_lds_b128)
      for (int it = 0; it < 32; ++it) {
        int off = (it * 128 + threadIdx.x) * 16;  // byte offset, 2KB per instruction
        __builtin_amdgcn_global_load_async_to_lds_b128(
            (AS1 v4i*)(src + off), (AS3 v4i*)((char*)Blds + off), 0, 0);
      }
      wait_async0();
#else
      for (int it = 0; it < 32; ++it) {
        int off = it * 128 + threadIdx.x;         // int4 index
        ((int4*)Blds)[off] = ((const int4*)src)[off];
      }
#endif
    }
    __syncthreads();

    for (int tt = 0; tt < 4; ++tt) {
      const int t = tsup * 4 + tt;
      float qv = qb2[(size_t)b * kH + t * 16 + nl];  // same n for all 8 C elems of lane
      v8f c;
#pragma unroll
      for (int j = 0; j < 8; ++j) c[j] = qv;
      for (int kc = 0; kc < 16; ++kc) {              // K chunks of 32
        v16bf a = *(const v16bf*)(Alds + ((size_t)wave * 16 + kc) * 512 + lane * 16);
        v16bf bm = *(const v16bf*)(Blds + ((size_t)(tt * 16 + kc)) * 512 + lane * 16);
        c = __builtin_amdgcn_wmma_f32_16x16x32_bf16(false, a, false, bm, (short)0, c,
                                                    false, false);
      }
      float uv = u1[t * 16 + nl];
#pragma unroll
      for (int j = 0; j < 8; ++j) part[j] += tanhf(c[j]) * uv;
    }
  }

  // reduce over the 16 n-lanes of each half (lanes 0-15: rows 0..7, 16-31: rows 8..15)
#pragma unroll
  for (int j = 0; j < 8; ++j)
    for (int m = 1; m <= 8; m <<= 1) part[j] += __shfl_xor(part[j], m, 32);

  if (nl == 0) {
    int mbase = kh * 8;
    float* sp = scores + (size_t)b * kL + l0 + mbase;
#pragma unroll
    for (int j = 0; j < 8; ++j) sp[j] = part[j];
  }
}

// ---------------- masked softmax over L per batch ----------------
__global__ void softmax_kernel(const float* __restrict__ scores, const int* __restrict__ mask,
                               float* __restrict__ attn) {
  __shared__ float red[256];
  const int b = blockIdx.x, tid = threadIdx.x;
  const float* sr = scores + (size_t)b * kL;
  const int* mr = mask + (size_t)b * kL;
  float m = -3.4e38f;
  for (int l = tid; l < kL; l += 256) if (mr[l]) m = fmaxf(m, sr[l]);
  red[tid] = m; __syncthreads();
  for (int s = 128; s > 0; s >>= 1) { if (tid < s) red[tid] = fmaxf(red[tid], red[tid + s]); __syncthreads(); }
  m = red[0]; __syncthreads();

  float sum = 0.0f;
  float* ar = attn + (size_t)b * kL;
  for (int l = tid; l < kL; l += 256) {
    float e = mr[l] ? __expf(sr[l] - m) : 0.0f;
    ar[l] = e; sum += e;
  }
  red[tid] = sum; __syncthreads();
  for (int s = 128; s > 0; s >>= 1) { if (tid < s) red[tid] += red[tid + s]; __syncthreads(); }
  float inv = 1.0f / red[0];
  for (int l = tid; l < kL; l += 256) ar[l] *= inv;
}

// ---------------- context = attn-weighted sum of document rows ----------------
__global__ void context_kernel(const float* __restrict__ attn, const float* __restrict__ doc,
                               float* __restrict__ ctx) {
  __shared__ float sa[256];
  const int b = blockIdx.x >> 1;
  const int hcol = ((blockIdx.x & 1) << 8) + threadIdx.x;  // 0..511
  float acc = 0.0f;
  for (int l0 = 0; l0 < kL; l0 += 256) {
    __syncthreads();
    sa[threadIdx.x] = attn[(size_t)b * kL + l0 + threadIdx.x];
    __syncthreads();
    const float* dp = doc + ((size_t)b * kL + l0) * kH + hcol;
    for (int i = 0; i < 256; ++i) {
      __builtin_prefetch(dp + (size_t)(i + 8) * kH, 0, 1);
      acc = fmaf(sa[i], dp[(size_t)i * kH], acc);
    }
  }
  ctx[(size_t)b * kH + hcol] = acc;
}

// ---------------- pack [h | context] (32 x 1024) into WMMA A-fragments (bf16) ------
// frag addr = (mt*32 + kc)*512 + lane*16 + slot ; K = kc*32 + (s>>3)*16 + kh*8 + (s&7)
__global__ void pack_combined_kernel(const float* __restrict__ h, const float* __restrict__ ctx,
                                     __bf16* __restrict__ frag) {
  int idx = blockIdx.x * blockDim.x + threadIdx.x;       // [0, 2*32*512)
  int s = idx & 15;
  int lane = (idx >> 4) & 31;
  int rem = idx >> 9;
  int kc = rem & 31, mt = rem >> 5;
  int m = lane & 15, kh = lane >> 4;
  int K = kc * 32 + ((s >> 3) << 4) + (kh << 3) + (s & 7);
  int b = mt * 16 + m;
  float val = (K < kH) ? h[(size_t)b * kH + K] : ctx[(size_t)b * kH + (K - kH)];
  frag[idx] = (__bf16)val;
}

// ---------------- output projection (WMMA bf16): out = [h|ctx] @ Wout^T + b ----------
// one wave per 16-col V tile; both 16-row M tiles computed so Wout is read once.
__global__ void __launch_bounds__(128)
outproj_kernel(const __bf16* __restrict__ afrag, const float* __restrict__ Woutw,
               const float* __restrict__ Woutb, float* __restrict__ out) {
  const int wave = threadIdx.x >> 5;
  const int lane = threadIdx.x & 31;
  const int nl = lane & 15, kh = lane >> 4;
  const int t = blockIdx.x * 4 + wave;                   // 2000 tiles exactly
  const int v0 = t * 16;

  float bias = Woutb[v0 + nl];
  v8f c0, c1;
#pragma unroll
  for (int j = 0; j < 8; ++j) { c0[j] = bias; c1[j] = bias; }

  for (int kc = 0; kc < 32; ++kc) {
    v16bf a0 = *(const v16bf*)&afrag[((size_t)0 * 32 + kc) * 512 + lane * 16];
    v16bf a1 = *(const v16bf*)&afrag[((size_t)1 * 32 + kc) * 512 + lane * 16];
    int K = kc * 32 + (kh << 3);                         // B lane mapping: lane -> N, slot -> K
    const float* wp = Woutw + (size_t)(v0 + nl) * (2 * kH) + K;  // row of Wout
    v16bf bm;
#pragma unroll
    for (int s = 0; s < 16; ++s) bm[s] = (__bf16)wp[((s >> 3) << 4) + (s & 7)];
    c0 = __builtin_amdgcn_wmma_f32_16x16x32_bf16(false, a0, false, bm, (short)0, c0,
                                                 false, false);
    c1 = __builtin_amdgcn_wmma_f32_16x16x32_bf16(false, a1, false, bm, (short)0, c1,
                                                 false, false);
  }

  const int v = v0 + nl;
#pragma unroll
  for (int j = 0; j < 8; ++j) {
    out[(size_t)(j + 8 * kh) * kV + v] = c0[j];          // batch rows 0..15
    out[(size_t)(16 + j + 8 * kh) * kV + v] = c1[j];     // batch rows 16..31
  }
}

} // namespace

extern "C" void kernel_launch(void* const* d_in, const int* in_sizes, int n_in,
                              void* d_out, int out_size, void* d_ws, size_t ws_size,
                              hipStream_t stream) {
  const float* x    = (const float*)d_in[0];
  const float* h0   = (const float*)d_in[1];
  const float* c0   = (const float*)d_in[2];
  const float* doc  = (const float*)d_in[3];
  const int*   mask = (const int*)d_in[4];
  const float* Wih  = (const float*)d_in[5];
  const float* Whh  = (const float*)d_in[6];
  const float* bih  = (const float*)d_in[7];
  const float* bhh  = (const float*)d_in[8];
  const float* W1w  = (const float*)d_in[9];
  const float* W1b  = (const float*)d_in[10];
  const float* W2w  = (const float*)d_in[11];
  const float* W2b  = (const float*)d_in[12];
  const float* u1   = (const float*)d_in[13];
  const float* Woutw = (const float*)d_in[14];
  const float* Woutb = (const float*)d_in[15];

  float* out   = (float*)d_out;                  // (B,V) logits
  float* h_out = out + (size_t)kB * kV;          // (B,H)
  float* c_out = h_out + (size_t)kB * kH;        // (B,H)

  // workspace layout (bytes)
  char* ws = (char*)d_ws;
  float*  gates   = (float*)(ws);                          // 32*2048 f32   = 256KB
  float*  qb2     = (float*)(ws + (256 << 10));            // 32*512  f32   =  64KB
  float*  ctx     = (float*)(ws + (320 << 10));            // 32*512  f32   =  64KB
  float*  scores  = (float*)(ws + (384 << 10));            // 32*4096 f32   = 512KB
  float*  attn    = (float*)(ws + (896 << 10));            // 32*4096 f32   = 512KB
  __bf16* w2frag  = (__bf16*)(ws + (1408 << 10));          // 512*16*32 bf16= 512KB
  __bf16* cfrag   = (__bf16*)(ws + (1920 << 10));          // 2*32*512 bf16 =  64KB

  gates_kernel<<<256, 256, 0, stream>>>(x, h0, Wih, Whh, bih, bhh, gates);
  hc_kernel<<<64, 256, 0, stream>>>(gates, c0, h_out, c_out);
  q_kernel<<<64, 256, 0, stream>>>(h_out, W1w, W1b, W2b, qb2);
  pack_w2_kernel<<<1024, 256, 0, stream>>>(W2w, w2frag);
  scores_kernel<<<2048, 128, 131072, stream>>>(doc, w2frag, qb2, u1, scores);
  softmax_kernel<<<kB, 256, 0, stream>>>(scores, mask, attn);
  context_kernel<<<kB * 2, 256, 0, stream>>>(attn, doc, ctx);
  pack_combined_kernel<<<128, 256, 0, stream>>>(h_out, ctx, cfrag);
  outproj_kernel<<<500, 128, 0, stream>>>(cfrag, Woutw, Woutb, out);
}
